// AttentionMechanism_42253888258616
// MI455X (gfx1250) — matrix operand order repaired and checked by reading, not compiled, not run on hardware
//
#include <hip/hip_runtime.h>

typedef __attribute__((ext_vector_type(16))) __bf16 v16bf;
typedef __attribute__((ext_vector_type(8)))  __bf16 v8bf;
typedef __attribute__((ext_vector_type(2)))  __bf16 v2bf;
typedef __attribute__((ext_vector_type(8)))  float   v8f;
typedef __attribute__((ext_vector_type(4)))  float   v4f;
typedef int v4i __attribute__((vector_size(16)));

#define AS1 __attribute__((address_space(1)))
#define AS3 __attribute__((address_space(3)))

#define B_  4
#define NQ_ 4096
#define NK_ 4096
#define D_  1024

#define QT      64      // queries per block
#define KT      32      // keys per iteration
#define THREADS 512     // 16 waves: (mq 0..3) x (wd 0..3)
#define DW      256     // D columns per wave

// pitches chosen for bank-conflict-free access (64 banks x 4B)
#define VS_PITCH_B   2096   // bytes/row of Vs  (+16B-granule XOR swizzle by row&15)
#define VST_PITCH_E  40     // bf16/row of VsT  (80B: 20 dwords, gcd(20,64)=4)
#define S_PITCH_F    36     // floats/row of S  (144B, 16B aligned, gcd(36,64)=4)
#define P_PITCH_E    40     // bf16/row of P    (80B)

// ---- dynamic LDS layout (bytes) ----
#define VRAW_OFF   0
#define VRAW_BYTES (KT * D_ * 4)                  // 131072 raw fp32 tile (swizzled granules)
#define VS_OFF     (VRAW_OFF + VRAW_BYTES)
#define VS_BYTES   (KT * VS_PITCH_B)              // 67072
#define VST_OFF    (VS_OFF + VS_BYTES)
#define VST_BYTES  (D_ * VST_PITCH_E * 2)         // 81920
#define S_OFF      (VST_OFF + VST_BYTES)
#define S_BYTES    (QT * S_PITCH_F * 4)           // 9216
#define P_OFF      (S_OFF + S_BYTES)
#define P_BYTES    (QT * P_PITCH_E * 2)           // 5120
#define RM_OFF     (P_OFF + P_BYTES)
#define RL_OFF     (RM_OFF + QT * 4)
#define RA_OFF     (RL_OFF + QT * 4)
#define SMEM_BYTES (RA_OFF + QT * 4)              // 295168

#if defined(__has_builtin)
#if __has_builtin(__builtin_amdgcn_global_load_async_to_lds_b128) && \
    __has_builtin(__builtin_amdgcn_s_wait_asynccnt)
#define USE_ASYNC 1
#endif
#endif
#ifndef USE_ASYNC
#define USE_ASYNC 0
#endif

__global__ __launch_bounds__(THREADS)
void attn_fa_wmma_kernel(const float* __restrict__ Qg,
                         const float* __restrict__ Vg,
                         float* __restrict__ Og)
{
    extern __shared__ __align__(128) char smem[];
    char*   vrawB = smem + VRAW_OFF;
    char*   vsB   = smem + VS_OFF;
    char*   vstB  = smem + VST_OFF;
    float*  sS    = (float*)(smem + S_OFF);
    char*   pB    = smem + P_OFF;
    float*  rowM  = (float*)(smem + RM_OFF);
    float*  rowL  = (float*)(smem + RL_OFF);
    float*  rowA  = (float*)(smem + RA_OFF);

    const int tid   = threadIdx.x;
    const int lane  = tid & 31;
    const int wave  = tid >> 5;
    const int l15   = lane & 15;
    const int hi    = lane >> 4;
    const int mq    = wave >> 2;     // 0..3 : 16-query sub-tile
    const int wd    = wave & 3;      // 0..3 : 256-wide D chunk
    const int dbase = wd * DW;

    const int bid = blockIdx.x;
    const int b   = bid / (NQ_ / QT);
    const int q0  = (bid % (NQ_ / QT)) * QT;

    const size_t vbase = (size_t)b * NK_ * D_;   // elements

    // ---------------- preload Q fragments (A matrices, 16x32 bf16) ----------------
    const size_t qrow = ((size_t)b * NQ_ + q0 + mq * 16 + l15) * D_;
    v16bf aQ[8];
#pragma unroll
    for (int c = 0; c < 8; ++c) {
        const float* src0 = Qg + qrow + dbase + c * 32 + 16 * hi;  // K = 16*hi + s
        const float* src1 = src0 + 8;                               // K = 16*hi + 8 + s
#pragma unroll
        for (int s = 0; s < 8; ++s) {
            aQ[c][s]     = (__bf16)src0[s];
            aQ[c][s + 8] = (__bf16)src1[s];
        }
    }

    if (tid < QT) {
        rowM[tid] = -3.0e38f;
        rowL[tid] = 0.0f;
    }

    v8f accC[16];
#pragma unroll
    for (int nt = 0; nt < 16; ++nt) accC[nt] = v8f{};

    // -------- tile fetch: global fp32 -> Vraw (16B granules XOR-swizzled by row&15) ----
    // wave covers rows 2*wave + (lane>>4); 16 b128 ops cover 256 granules per row.
    const int frow = 2 * wave + hi;
    const int fsw  = frow & 15;

#define ISSUE_TILE(K0N)                                                               \
    do {                                                                              \
        const char* gsrc = (const char*)(Vg + vbase + (size_t)((K0N) + frow) * D_);   \
        char*       ldst = vrawB + frow * 4096;                                       \
        _Pragma("unroll")                                                             \
        for (int u = 0; u < 16; ++u) {                                                \
            const int gg = u * 16 + l15;                                              \
            FETCH16(gsrc + gg * 16, ldst + ((gg ^ fsw) << 4));                        \
        }                                                                             \
    } while (0)

#if USE_ASYNC
#define FETCH16(GP, LP)                                                               \
    __builtin_amdgcn_global_load_async_to_lds_b128((AS1 v4i*)(char*)(GP),             \
                                                   (AS3 v4i*)(LP), 0, 0)
#define WAIT_TILE() __builtin_amdgcn_s_wait_asynccnt(0)
#else
#define FETCH16(GP, LP) *(v4f*)(LP) = *(const v4f*)(GP)
#define WAIT_TILE() ((void)0)
#endif

    ISSUE_TILE(0);   // prologue: start fetching tile 0

    for (int k0 = 0; k0 < NK_; k0 += KT) {
        WAIT_TILE();            // my async writes to Vraw have landed
        __syncthreads();        // everyone's writes landed; all readers of Vs/VsT/P done

        // -------- conversion pass A: Vraw -> Vs (row-major bf16, swizzled) --------
        {
            const int rowA_ = tid >> 4;          // 0..31
            const int swA   = rowA_ & 15;
#pragma unroll
            for (int u = 0; u < 8; ++u) {
                const int g  = (tid & 15) + 16 * u;   // 16B bf16-granule (8 cols)
                const int ga = g * 2;                 // 16B fp32-granule
                v4f f0 = *(const v4f*)(vrawB + rowA_ * 4096 + (((ga    ) ^ swA) << 4));
                v4f f1 = *(const v4f*)(vrawB + rowA_ * 4096 + (((ga + 1) ^ swA) << 4));
                v8bf o;
                o[0] = (__bf16)f0[0]; o[1] = (__bf16)f0[1];
                o[2] = (__bf16)f0[2]; o[3] = (__bf16)f0[3];
                o[4] = (__bf16)f1[0]; o[5] = (__bf16)f1[1];
                o[6] = (__bf16)f1[2]; o[7] = (__bf16)f1[3];
                *(v8bf*)(vsB + rowA_ * VS_PITCH_B + ((g ^ swA) << 4)) = o;
            }
        }
        // -------- conversion pass B: Vraw -> VsT (transposed bf16, padded pitch) --------
        {
            const int rp  = tid & 15;            // row pair: rows 2rp, 2rp+1
            const int d00 = (tid >> 4) * 32;     // 32 D-columns
            const int r0 = 2 * rp, r1 = 2 * rp + 1;
            v2bf* vst2 = (v2bf*)vstB;
#pragma unroll
            for (int u = 0; u < 8; ++u) {
                const int d0 = d00 + 4 * u;
                const int gb = d0 >> 2;
                v4f a0 = *(const v4f*)(vrawB + r0 * 4096 + ((gb ^ (r0 & 15)) << 4));
                v4f a1 = *(const v4f*)(vrawB + r1 * 4096 + ((gb ^ (r1 & 15)) << 4));
#pragma unroll
                for (int e = 0; e < 4; ++e) {
                    v2bf p2; p2[0] = (__bf16)a0[e]; p2[1] = (__bf16)a1[e];
                    vst2[(d0 + e) * (VST_PITCH_E / 2) + rp] = p2;
                }
            }
        }
        // zero the score buffer (8 threads x 4 floats per row)
        *(v4f*)(sS + (tid >> 3) * S_PITCH_F + (tid & 7) * 4) = v4f{};
        __syncthreads();

        // overlap: start fetching the NEXT tile while we compute on this one
        if (k0 + KT < NK_) ISSUE_TILE(k0 + KT);

        // -------- scores: partial S = Q_chunk @ K_chunk^T, reduce via ds_add_f32 --------
        v8f accS0 = v8f{};
        v8f accS1 = v8f{};
#pragma unroll
        for (int c = 0; c < 8; ++c) {
            const int g0 = (dbase + c * 32 + 16 * hi) >> 3;  // 16B granule in Vs row
            // keys 0..15 (r = l15) and 16..31 (r = 16+l15); r&15 == l15 for both
            const char* kr0 = vsB + (l15)      * VS_PITCH_B;
            const char* kr1 = vsB + (16 + l15) * VS_PITCH_B;
            v8bf b0l = *(const v8bf*)(kr0 + (((g0    ) ^ l15) << 4));
            v8bf b0h = *(const v8bf*)(kr0 + (((g0 + 1) ^ l15) << 4));
            v8bf b1l = *(const v8bf*)(kr1 + (((g0    ) ^ l15) << 4));
            v8bf b1h = *(const v8bf*)(kr1 + (((g0 + 1) ^ l15) << 4));
            v16bf b0 = __builtin_shufflevector(b0l, b0h, 0,1,2,3,4,5,6,7,8,9,10,11,12,13,14,15);
            v16bf b1 = __builtin_shufflevector(b1l, b1h, 0,1,2,3,4,5,6,7,8,9,10,11,12,13,14,15);
            accS0 = __builtin_amdgcn_wmma_f32_16x16x32_bf16(false, aQ[c], false, b0,
                                                            (short)0, accS0, false, false);
            accS1 = __builtin_amdgcn_wmma_f32_16x16x32_bf16(false, aQ[c], false, b1,
                                                            (short)0, accS1, false, false);
        }
#pragma unroll
        for (int j = 0; j < 8; ++j) {
            const int r = mq * 16 + j + 8 * hi;
            atomicAdd(&sS[r * S_PITCH_F + l15],      accS0[j]);
            atomicAdd(&sS[r * S_PITCH_F + 16 + l15], accS1[j]);
        }
        __syncthreads();

        // -------- online softmax (fp32), one thread per query row --------
        if (tid < QT) {
            const int r = tid;
            const float mOld = rowM[r];
            float mNew = mOld;
#pragma unroll
            for (int j = 0; j < KT; ++j) mNew = fmaxf(mNew, sS[r * S_PITCH_F + j]);
            const float alpha = __expf(mOld - mNew);
            float l = rowL[r] * alpha;
            __bf16* pRow = (__bf16*)(pB + r * P_PITCH_E * 2);
#pragma unroll
            for (int j = 0; j < KT; ++j) {
                const float p = __expf(sS[r * S_PITCH_F + j] - mNew);
                l += p;
                pRow[j] = (__bf16)p;
            }
            rowM[r] = mNew;
            rowL[r] = l;
            rowA[r] = alpha;
        }
        __syncthreads();

        // -------- context: acc = alpha*acc + P @ V_tile --------
        float a[8];
#pragma unroll
        for (int j = 0; j < 8; ++j) a[j] = rowA[mq * 16 + 8 * hi + j];

        const char* prow = pB + (mq * 16 + l15) * (P_PITCH_E * 2);
        v8bf plo = *(const v8bf*)(prow + 32 * hi);          // K = 16*hi + s
        v8bf phi = *(const v8bf*)(prow + 32 * hi + 16);     // K = 16*hi + 8 + s
        v16bf aP = __builtin_shufflevector(plo, phi,
                                           0,1,2,3,4,5,6,7,8,9,10,11,12,13,14,15);
#pragma unroll
        for (int nt = 0; nt < 16; ++nt) {
#pragma unroll
            for (int j = 0; j < 8; ++j) accC[nt][j] *= a[j];
            const int dcol = dbase + nt * 16 + l15;
            const char* vb = vstB + dcol * (VST_PITCH_E * 2) + 32 * hi;
            v8bf blo = *(const v8bf*)(vb);
            v8bf bhi = *(const v8bf*)(vb + 16);
            v16bf bV = __builtin_shufflevector(blo, bhi,
                                               0,1,2,3,4,5,6,7,8,9,10,11,12,13,14,15);
            accC[nt] = __builtin_amdgcn_wmma_f32_16x16x32_bf16(false, aP, false, bV,
                                                               (short)0, accC[nt], false, false);
        }
        // next loop top's WAIT_TILE + __syncthreads protects Vs/VsT/S/P
    }

    // -------- epilogue: divide by running sum, store fp32 --------
    float linv[8];
#pragma unroll
    for (int j = 0; j < 8; ++j) linv[j] = 1.0f / rowL[mq * 16 + 8 * hi + j];

#pragma unroll
    for (int nt = 0; nt < 16; ++nt) {
#pragma unroll
        for (int j = 0; j < 8; ++j) {
            const int r = mq * 16 + 8 * hi + j;
            const size_t o = ((size_t)b * NQ_ + q0 + r) * D_ + dbase + nt * 16 + l15;
            Og[o] = accC[nt][j] * linv[j];
        }
    }
}

extern "C" void kernel_launch(void* const* d_in, const int* in_sizes, int n_in,
                              void* d_out, int out_size, void* d_ws, size_t ws_size,
                              hipStream_t stream) {
    const float* Qg = (const float*)d_in[0];
    const float* Vg = (const float*)d_in[1];
    float*       Og = (float*)d_out;

    (void)hipFuncSetAttribute(reinterpret_cast<const void*>(attn_fa_wmma_kernel),
                              hipFuncAttributeMaxDynamicSharedMemorySize, SMEM_BYTES);

    dim3 grid(B_ * (NQ_ / QT));   // 256 blocks
    dim3 block(THREADS);          // 512 threads = 16 wave32
    attn_fa_wmma_kernel<<<grid, block, SMEM_BYTES, stream>>>(Qg, Vg, Og);
}
